// QuantizedTranscriptformer_30288109371667
// MI455X (gfx1250) — matrix operand order repaired and checked by reading, not compiled, not run on hardware
//
#include <hip/hip_runtime.h>
#include <hip/hip_bf16.h>
#include <hip/hip_fp16.h>

// MI455X / gfx1250, wave32, WMMA f32_16x16x32_f16 for all GEMMs.
// Async global->LDS staging (ASYNCcnt) where the toolchain exposes it.

typedef _Float16 h16;
typedef __attribute__((ext_vector_type(16))) _Float16 v16h;
typedef __attribute__((ext_vector_type(8)))  float    v8f;

static constexpr int BB   = 4;
static constexpr int SS   = 1024;
static constexpr int DD   = 512;
static constexpr int HH   = 8;
static constexpr int DK   = 64;
static constexpr int LL   = 4;
static constexpr int DFF  = 2048;
static constexpr int NT   = BB * SS;          // 4096 token rows

// ---------- async global->LDS (guarded; falls back to manual staging) ----------

#if __has_builtin(__builtin_amdgcn_global_load_async_to_lds_b128)
#define HAVE_ASYNC_LDS 1
typedef int v4i __attribute__((vector_size(16)));   // matches builtin pointee type
__device__ __forceinline__ void async_copy16(const h16* g, h16* l) {
  // per-lane 16B: LDS[l] = MEM[g], no VGPR data, tracked by ASYNCcnt
  __builtin_amdgcn_global_load_async_to_lds_b128((v4i*)g, (v4i*)l, 0, 0);
}
__device__ __forceinline__ void async_wait0() {
#if __has_builtin(__builtin_amdgcn_s_wait_asynccnt)
  __builtin_amdgcn_s_wait_asynccnt(0);
#else
  asm volatile("s_wait_asynccnt 0" ::: "memory");
#endif
}
#else
#define HAVE_ASYNC_LDS 0
#endif

// ---------- helpers ----------

__device__ __forceinline__ v8f zero8() {
  v8f z;
#pragma unroll
  for (int i = 0; i < 8; ++i) z[i] = 0.0f;
  return z;
}

// load 16 halves (8 from p0, 8 from p1) into a WMMA fragment register set
__device__ __forceinline__ v16h ld16(const h16* p0, const h16* p1) {
  v16h r;
#pragma unroll
  for (int i = 0; i < 8; ++i) { r[i] = p0[i]; r[i + 8] = p1[i]; }
  return r;
}

__device__ __forceinline__ float gelu_tanh(float x) {
  return 0.5f * x * (1.0f + tanhf(0.7978845608028654f * (x + 0.044715f * x * x * x)));
}

// ---------- weight f32 -> f16 convert ----------

__global__ void __launch_bounds__(256) cvt_kernel(const float* __restrict__ in,
                                                  h16* __restrict__ out, int n) {
  int i = blockIdx.x * 256 + threadIdx.x;
  if (i < n) out[i] = (h16)in[i];
}

// ---------- embedding + shifted counts/keep ----------

__global__ void __launch_bounds__(256) embed_kernel(const int* __restrict__ tok,
    const float* __restrict__ cnt, const float* __restrict__ emb,
    float* __restrict__ x, float* __restrict__ biasv, float* __restrict__ keepf) {
  const int bs = blockIdx.x, tid = threadIdx.x;
  const int b = bs >> 10, s = bs & 1023;
  const int t = (s == 0) ? 1 : tok[b * SS + s - 1];          // START_IDX = 1
  x[(size_t)bs * DD + tid]        = emb[(size_t)t * DD + tid];
  x[(size_t)bs * DD + 256 + tid]  = emb[(size_t)t * DD + 256 + tid];
  if (tid == 0) {
    float c = (s == 0) ? 1.0f : cnt[b * SS + s - 1];
    biasv[bs] = log1pf(c + 1e-6f);                           // log1p(counts + eps)
    keepf[bs] = (t != 0) ? 1.0f : 0.0f;                      // PAD_IDX = 0
  }
}

// ---------- layernorm (writes f32 + f16 copies) ----------

__global__ void __launch_bounds__(256) ln_kernel(const float* __restrict__ x,
    const float* __restrict__ g, const float* __restrict__ bt,
    float* __restrict__ xn, h16* __restrict__ xnh) {
  __shared__ float red[256];
  const int row = blockIdx.x, tid = threadIdx.x;
  const float v0 = x[(size_t)row * DD + tid];
  const float v1 = x[(size_t)row * DD + 256 + tid];
  red[tid] = v0 + v1; __syncthreads();
#pragma unroll
  for (int off = 128; off >= 1; off >>= 1) {
    if (tid < off) red[tid] += red[tid + off];
    __syncthreads();
  }
  const float mu = red[0] * (1.0f / 512.0f); __syncthreads();
  const float d0 = v0 - mu, d1 = v1 - mu;
  red[tid] = d0 * d0 + d1 * d1; __syncthreads();
#pragma unroll
  for (int off = 128; off >= 1; off >>= 1) {
    if (tid < off) red[tid] += red[tid + off];
    __syncthreads();
  }
  const float rs = rsqrtf(red[0] * (1.0f / 512.0f) + 1e-5f); __syncthreads();
  const float y0 = d0 * rs * g[tid] + bt[tid];
  const float y1 = d1 * rs * g[256 + tid] + bt[256 + tid];
  xn[(size_t)row * DD + tid]        = y0;
  xn[(size_t)row * DD + 256 + tid]  = y1;
  xnh[(size_t)row * DD + tid]       = (h16)y0;
  xnh[(size_t)row * DD + 256 + tid] = (h16)y1;
}

// ---------- TurboQuant quant->dequant, in place on f16 BHSD vectors of 64 ----------

__device__ __forceinline__ float grp_sum64(float v, float* red, int tid) {
  red[tid] = v; __syncthreads();
#pragma unroll
  for (int off = 32; off >= 1; off >>= 1) {
    if ((tid & 63) < off) red[tid] += red[tid + off];
    __syncthreads();
  }
  float r = red[tid & ~63]; __syncthreads();
  return r;
}

__global__ void __launch_bounds__(256) quant_kernel(h16* __restrict__ kv,
                                                    const float* __restrict__ cb) {
  __shared__ float red[256];
  const int tid = threadIdx.x;
  const size_t vec = (size_t)blockIdx.x * 4 + (tid >> 6);
  const int li = tid & 63;
  float lev[8];
#pragma unroll
  for (int j = 0; j < 8; ++j) lev[j] = cb[j];
  const float e = (float)kv[vec * 64 + li];
  const float ss = grp_sum64(e * e, red, tid);
  const float nrm = sqrtf(ss);
  const float u = e / (nrm + 1e-8f);
  const float a = fabsf(u);
  const float sgn = (u >= 0.0f) ? 1.0f : -1.0f;
  float q = lev[0], bd = fabsf(a - lev[0]);
#pragma unroll
  for (int j = 1; j < 8; ++j) {
    float d = fabsf(a - lev[j]);
    if (d < bd) { bd = d; q = lev[j]; }
  }
  const float num = grp_sum64(a * q, red, tid);
  const float den = grp_sum64(q * q, red, tid);
  const float rmag = num / (den + 1e-8f);
  kv[vec * 64 + li] = (h16)(sgn * q * rmag * nrm);
}

// ---------- WMMA GEMM: C[M,N] = A[M,K] * B[K,N] + bias, templated epilogue ----------
// MODE 0: write f16 to [B,H,S,DK] layout (Q/K/V projections)
// MODE 1: write f32 row-major + residual add (Wo proj, FFN2)
// MODE 2: write f16 row-major with tanh-GELU (FFN1)

template<int MODE>
__global__ void __launch_bounds__(256) gemm_kernel(const h16* __restrict__ A,
    const h16* __restrict__ Bw, const float* __restrict__ bias,
    float* __restrict__ outf, h16* __restrict__ outh,
    const float* __restrict__ resid, int M, int N, int K) {
  __shared__ __align__(16) h16 As[64 * 32];     // A tile, row-major [row][k]
  __shared__ __align__(16) h16 Bst[64 * 32];    // B tile, transposed [col][k]
  const int tid  = threadIdx.x;
  const int lane = tid & 31, wv = tid >> 5;
  const int wr = wv & 3, wc = wv >> 2;          // 4x2 wave grid -> 64x64 tile
  const int m = lane & 15, hi = lane >> 4;
  const int rb = blockIdx.y * 64, cb = blockIdx.x * 64;

  v8f c0 = zero8(), c1 = zero8();

  for (int k0 = 0; k0 < K; k0 += 32) {
    {   // stage A: 64 rows x 32 k (16B per thread)
      const int r = tid >> 2, cg = (tid & 3) * 8;
      const h16* src = A + (size_t)(rb + r) * K + k0 + cg;
#if HAVE_ASYNC_LDS
      async_copy16(src, &As[r * 32 + cg]);      // global_load_async_to_lds_b128
#else
#pragma unroll
      for (int i = 0; i < 8; ++i) As[r * 32 + cg + i] = src[i];
#endif
    }
    {   // stage B transposed: 32 k x 64 cols -> Bst[col][k] (keeps frag loads contiguous)
      const int r = tid >> 3, cg = (tid & 7) * 8;
      const h16* src = Bw + (size_t)(k0 + r) * N + cb + cg;
#pragma unroll
      for (int i = 0; i < 8; ++i) Bst[(cg + i) * 32 + r] = src[i];
    }
    if (k0 + 32 < K) {  // global_prefetch_b8 for next weight tile
      __builtin_prefetch(Bw + (size_t)(k0 + 32 + (tid >> 3)) * N + cb, 0, 1);
    }
#if HAVE_ASYNC_LDS
    async_wait0();
#endif
    __syncthreads();

    // A fragment: M = lane%16, K = (j>=8?16:0) + j%8 + 8*hi
    const int arow = wr * 16 + m;
    v16h af = ld16(&As[arow * 32 + hi * 8], &As[arow * 32 + 16 + hi * 8]);
    // B fragments: N = lane%16 (+16), K = (j>=8?16:0) + j%8 + 8*hi
    const int bc0 = wc * 32 + m;
    v16h bf0 = ld16(&Bst[bc0 * 32 + hi * 8],        &Bst[bc0 * 32 + 16 + hi * 8]);
    v16h bf1 = ld16(&Bst[(bc0 + 16) * 32 + hi * 8], &Bst[(bc0 + 16) * 32 + 16 + hi * 8]);

    c0 = __builtin_amdgcn_wmma_f32_16x16x32_f16(false, af, false, bf0, (short)0, c0, false, false);
    c1 = __builtin_amdgcn_wmma_f32_16x16x32_f16(false, af, false, bf1, (short)0, c1, false, false);
    __syncthreads();
  }

  // epilogue: C element (t,r): row = rb + wr*16 + 8*hi + r, col = cb + wc*32 + 16*t + m
#pragma unroll
  for (int t = 0; t < 2; ++t) {
    v8f cc = t ? c1 : c0;
#pragma unroll
    for (int r = 0; r < 8; ++r) {
      const int row = rb + wr * 16 + hi * 8 + r;
      const int col = cb + wc * 32 + t * 16 + m;
      float v = cc[r] + bias[col];
      if (MODE == 0) {
        const int b = row >> 10, s = row & 1023, hh = col >> 6, dk = col & 63;
        outh[(((size_t)(b * HH + hh) << 10) + s) * DK + dk] = (h16)v;
      } else if (MODE == 1) {
        outf[(size_t)row * N + col] = v + resid[(size_t)row * N + col];
      } else {
        outh[(size_t)row * N + col] = (h16)gelu_tanh(v);
      }
    }
  }
}

// ---------- flash attention: 1 wave per 16-query tile, 32-key blocks ----------

__global__ void __launch_bounds__(32) attn_kernel(const h16* __restrict__ Q,
    const h16* __restrict__ Kq, const h16* __restrict__ Vq,
    const float* __restrict__ biasv, const float* __restrict__ keepf,
    h16* __restrict__ O) {
  __shared__ __align__(16) h16 Kt[32 * 64];     // keys row-major [key][d]
  __shared__ __align__(16) h16 VtT[64 * 32];    // V transposed   [d][key]
  __shared__ __align__(16) h16 Pt[16 * 32];     // probs          [q][key]
  const int lane = threadIdx.x;
  const int m = lane & 15, hi = lane >> 4;
  const int qt = blockIdx.x & 63;               // S/16 = 64 q-tiles
  const int bh = blockIdx.x >> 6;
  const int b = bh >> 3, h = bh & 7;
  const int qbase = qt * 16;
  const size_t base = (size_t)bh * SS * DK;

  // Q fragments held in registers for the whole pass (16 x 64)
  const h16* qrow = Q + base + (size_t)(qbase + m) * DK;
  const v16h a0 = ld16(qrow + hi * 8,      qrow + 16 + hi * 8);   // d 0..31
  const v16h a1 = ld16(qrow + 32 + hi * 8, qrow + 48 + hi * 8);   // d 32..63

  v8f o0 = zero8(), o1 = zero8(), o2 = zero8(), o3 = zero8();
  float mrun[8], lrun[8];
#pragma unroll
  for (int r = 0; r < 8; ++r) { mrun[r] = -3.0e38f; lrun[r] = 0.0f; }
  const float scale = 0.125f;                   // 1/sqrt(64)
  const float icap  = 1.0f / 30.0f;

  const int kbmax = (qbase + 15) >> 5;          // causal upper bound
  for (int kb = 0; kb <= kbmax; ++kb) {
    const int kbase = kb * 32;
    const h16* ksrc = Kq + base + (size_t)(kbase + lane) * DK;
    const h16* vsrc = Vq + base + (size_t)(kbase + lane) * DK;
#if HAVE_ASYNC_LDS
#pragma unroll
    for (int c = 0; c < 8; ++c)                 // K row-major via async DMA
      async_copy16(ksrc + c * 8, &Kt[lane * 64 + c * 8]);
#else
#pragma unroll
    for (int d = 0; d < 64; ++d) Kt[lane * 64 + d] = ksrc[d];
#endif
#pragma unroll
    for (int d = 0; d < 64; ++d) VtT[d * 32 + lane] = vsrc[d];   // V transposed
#if HAVE_ASYNC_LDS
    async_wait0();
#endif
    __syncthreads();

    // scores S = Q(16x64) x K^T(64x16) per 16-key tile; B-frag from Kt rows
    v8f s0 = zero8(), s1 = zero8();
    {
      v16h b00 = ld16(&Kt[m * 64 + hi * 8],        &Kt[m * 64 + 16 + hi * 8]);
      v16h b01 = ld16(&Kt[m * 64 + 32 + hi * 8],   &Kt[m * 64 + 48 + hi * 8]);
      v16h b10 = ld16(&Kt[(16 + m) * 64 + hi * 8], &Kt[(16 + m) * 64 + 16 + hi * 8]);
      v16h b11 = ld16(&Kt[(16 + m) * 64 + 32 + hi * 8], &Kt[(16 + m) * 64 + 48 + hi * 8]);
      s0 = __builtin_amdgcn_wmma_f32_16x16x32_f16(false, a0, false, b00, (short)0, s0, false, false);
      s0 = __builtin_amdgcn_wmma_f32_16x16x32_f16(false, a1, false, b01, (short)0, s0, false, false);
      s1 = __builtin_amdgcn_wmma_f32_16x16x32_f16(false, a0, false, b10, (short)0, s1, false, false);
      s1 = __builtin_amdgcn_wmma_f32_16x16x32_f16(false, a1, false, b11, (short)0, s1, false, false);
    }

    const int key0 = kbase + m, key1 = key0 + 16;
    const float bi0 = biasv[b * SS + key0], bi1 = biasv[b * SS + key1];
    const bool  kp0 = keepf[b * SS + key0] > 0.5f;
    const bool  kp1 = keepf[b * SS + key1] > 0.5f;

#pragma unroll
    for (int r = 0; r < 8; ++r) {
      const int qg = qbase + hi * 8 + r;
      float x0 = tanhf((s0[r] * scale + bi0) * icap) * 30.0f;
      float x1 = tanhf((s1[r] * scale + bi1) * icap) * 30.0f;
      if (key0 > qg || !kp0) x0 = -3.0e38f;
      if (key1 > qg || !kp1) x1 = -3.0e38f;
      // online softmax; 16-lane half-wave reduction leaves row stats replicated
      float mx = fmaxf(x0, x1);
#pragma unroll
      for (int off = 8; off >= 1; off >>= 1) mx = fmaxf(mx, __shfl_xor(mx, off, 16));
      const float mnew = fmaxf(mrun[r], mx);
      const float corr = __expf(mrun[r] - mnew);
      const float p0 = __expf(x0 - mnew);
      const float p1 = __expf(x1 - mnew);
      float ps = p0 + p1;
#pragma unroll
      for (int off = 8; off >= 1; off >>= 1) ps += __shfl_xor(ps, off, 16);
      lrun[r] = lrun[r] * corr + ps;
      mrun[r] = mnew;
      o0[r] *= corr; o1[r] *= corr; o2[r] *= corr; o3[r] *= corr;
      Pt[(hi * 8 + r) * 32 + m]      = (h16)p0;
      Pt[(hi * 8 + r) * 32 + 16 + m] = (h16)p1;
    }
    __syncthreads();

    // O(16x64) += P(16x32) x V(32x64); A-frag from Pt, B-frags from VtT rows
    const v16h pa = ld16(&Pt[m * 32 + hi * 8], &Pt[m * 32 + 16 + hi * 8]);
    {
      v16h vb0 = ld16(&VtT[(0 * 16 + m) * 32 + hi * 8], &VtT[(0 * 16 + m) * 32 + 16 + hi * 8]);
      v16h vb1 = ld16(&VtT[(1 * 16 + m) * 32 + hi * 8], &VtT[(1 * 16 + m) * 32 + 16 + hi * 8]);
      v16h vb2 = ld16(&VtT[(2 * 16 + m) * 32 + hi * 8], &VtT[(2 * 16 + m) * 32 + 16 + hi * 8]);
      v16h vb3 = ld16(&VtT[(3 * 16 + m) * 32 + hi * 8], &VtT[(3 * 16 + m) * 32 + 16 + hi * 8]);
      o0 = __builtin_amdgcn_wmma_f32_16x16x32_f16(false, pa, false, vb0, (short)0, o0, false, false);
      o1 = __builtin_amdgcn_wmma_f32_16x16x32_f16(false, pa, false, vb1, (short)0, o1, false, false);
      o2 = __builtin_amdgcn_wmma_f32_16x16x32_f16(false, pa, false, vb2, (short)0, o2, false, false);
      o3 = __builtin_amdgcn_wmma_f32_16x16x32_f16(false, pa, false, vb3, (short)0, o3, false, false);
    }
    __syncthreads();
  }

  // epilogue: normalize by row sum, write o as [B,S, H*DK] f16 (GEMM input for Wo)
#pragma unroll
  for (int t = 0; t < 4; ++t) {
    v8f oc = (t == 0) ? o0 : (t == 1) ? o1 : (t == 2) ? o2 : o3;
#pragma unroll
    for (int r = 0; r < 8; ++r) {
      const int rowg = qbase + hi * 8 + r;
      const int col = h * DK + t * 16 + m;
      const float denom = fmaxf(lrun[r], 1e-30f);
      O[((size_t)b * SS + rowg) * DD + col] = (h16)(oc[r] / denom);
    }
  }
}

// ---------- host orchestration ----------

extern "C" void kernel_launch(void* const* d_in, const int* in_sizes, int n_in,
                              void* d_out, int out_size, void* d_ws, size_t ws_size,
                              hipStream_t stream) {
  (void)in_sizes; (void)n_in; (void)out_size; (void)ws_size;
  const int*   tok  = (const int*)d_in[0];
  const float* cnt  = (const float*)d_in[1];
  const float* emb  = (const float*)d_in[2];
  const float* Wq   = (const float*)d_in[3];
  const float* bq   = (const float*)d_in[4];
  const float* Wk   = (const float*)d_in[5];
  const float* bk   = (const float*)d_in[6];
  const float* Wv   = (const float*)d_in[7];
  const float* bv   = (const float*)d_in[8];
  const float* Wo   = (const float*)d_in[9];
  const float* bo   = (const float*)d_in[10];
  const float* ln1g = (const float*)d_in[11];
  const float* ln1b = (const float*)d_in[12];
  const float* ln2g = (const float*)d_in[13];
  const float* ln2b = (const float*)d_in[14];
  const float* W1   = (const float*)d_in[15];
  const float* b1   = (const float*)d_in[16];
  const float* W2   = (const float*)d_in[17];
  const float* b2   = (const float*)d_in[18];
  const float* cbk  = (const float*)d_in[19];

  char* p = (char*)d_ws;
  auto alloc = [&](size_t bytes) -> void* {
    char* r = p; p += (bytes + 255) & ~(size_t)255; return (void*)r;
  };

  h16*   wq16  = (h16*)alloc((size_t)LL * DD * DD * 2);
  h16*   wk16  = (h16*)alloc((size_t)LL * DD * DD * 2);
  h16*   wv16  = (h16*)alloc((size_t)LL * DD * DD * 2);
  h16*   wo16  = (h16*)alloc((size_t)LL * DD * DD * 2);
  h16*   w116  = (h16*)alloc((size_t)LL * DD * DFF * 2);
  h16*   w216  = (h16*)alloc((size_t)LL * DFF * DD * 2);
  float* x     = (float*)alloc((size_t)NT * DD * 4);
  float* xn    = (float*)alloc((size_t)NT * DD * 4);
  h16*   xn16  = (h16*)alloc((size_t)NT * DD * 2);
  h16*   q16   = (h16*)alloc((size_t)BB * HH * SS * DK * 2);
  h16*   k16   = (h16*)alloc((size_t)BB * HH * SS * DK * 2);
  h16*   v16b  = (h16*)alloc((size_t)BB * HH * SS * DK * 2);
  h16*   o16   = (h16*)alloc((size_t)NT * DD * 2);
  h16*   hbuf  = (h16*)alloc((size_t)NT * DFF * 2);
  float* biasv = (float*)alloc((size_t)NT * 4);
  float* keepf = (float*)alloc((size_t)NT * 4);

  auto cvt = [&](const float* in, h16* out, size_t n) {
    cvt_kernel<<<(unsigned)((n + 255) / 256), 256, 0, stream>>>(in, out, (int)n);
  };
  cvt(Wq, wq16, (size_t)LL * DD * DD);
  cvt(Wk, wk16, (size_t)LL * DD * DD);
  cvt(Wv, wv16, (size_t)LL * DD * DD);
  cvt(Wo, wo16, (size_t)LL * DD * DD);
  cvt(W1, w116, (size_t)LL * DD * DFF);
  cvt(W2, w216, (size_t)LL * DFF * DD);

  embed_kernel<<<NT, 256, 0, stream>>>(tok, cnt, emb, x, biasv, keepf);

  const dim3 gD(DD / 64, NT / 64);
  const dim3 gF(DFF / 64, NT / 64);
  for (int l = 0; l < LL; ++l) {
    ln_kernel<<<NT, 256, 0, stream>>>(x, ln1g + l * DD, ln1b + l * DD, xn, xn16);
    gemm_kernel<0><<<gD, 256, 0, stream>>>(xn16, wq16 + (size_t)l * DD * DD, bq + l * DD,
                                           nullptr, q16, nullptr, NT, DD, DD);
    gemm_kernel<0><<<gD, 256, 0, stream>>>(xn16, wk16 + (size_t)l * DD * DD, bk + l * DD,
                                           nullptr, k16, nullptr, NT, DD, DD);
    gemm_kernel<0><<<gD, 256, 0, stream>>>(xn16, wv16 + (size_t)l * DD * DD, bv + l * DD,
                                           nullptr, v16b, nullptr, NT, DD, DD);
    quant_kernel<<<(BB * HH * SS) / 4, 256, 0, stream>>>(k16, cbk + (size_t)(l * 2 + 0) * 8);
    quant_kernel<<<(BB * HH * SS) / 4, 256, 0, stream>>>(v16b, cbk + (size_t)(l * 2 + 1) * 8);
    attn_kernel<<<BB * HH * (SS / 16), 32, 0, stream>>>(q16, k16, v16b, biasv, keepf, o16);
    gemm_kernel<1><<<gD, 256, 0, stream>>>(o16, wo16 + (size_t)l * DD * DD, bo + l * DD,
                                           x, nullptr, xn, NT, DD, DD);
    ln_kernel<<<NT, 256, 0, stream>>>(x, ln2g + l * DD, ln2b + l * DD, xn, xn16);
    gemm_kernel<2><<<gF, 256, 0, stream>>>(xn16, w116 + (size_t)l * DD * DFF, b1 + l * DFF,
                                           nullptr, hbuf, nullptr, NT, DFF, DD);
    gemm_kernel<1><<<gD, 256, 0, stream>>>(hbuf, w216 + (size_t)l * DFF * DD, b2 + l * DD,
                                           x, nullptr, xn, NT, DD, DFF);
  }

  (void)hipMemcpyAsync(d_out, x, (size_t)NT * DD * sizeof(float),
                       hipMemcpyDeviceToDevice, stream);
}
